// CSBrainAlign_3753801417222
// MI455X (gfx1250) — compile-verified, matrix operand order checked by reading, not run on hardware
//
#include <hip/hip_runtime.h>
#include <hip/hip_bf16.h>

typedef __attribute__((ext_vector_type(16))) __bf16 v16bf;
typedef __attribute__((ext_vector_type(8)))  float  v8f;

#define BCv 128
#define Tv  384
#define Dv  128
#define DIv 256
#define MT  4      // 16-row tiles per wave (wave computes 64 x 16 output)

__device__ __forceinline__ float gelu_t(float x){
    float x3 = x*x*x;
    return 0.5f*x*(1.f + tanhf(0.79788456080286535588f*(x + 0.044715f*x3)));
}
__device__ __forceinline__ float silu_f(float x){ return x/(1.f + __expf(-x)); }
__device__ __forceinline__ float softplus_f(float x){ return (x > 20.f) ? x : log1pf(__expf(x)); }

// ---------------------------------------------------------------------------
// Fragment index mapping (ISA 7.12.2, 16-bit operands):
//   lane = half*16 + r   (r = M-row for A, N-col for B; same for both halves)
//   element e = 2j+c holds k = kt*32 + (j>>2)*16 + half*8 + (j&3)*2 + c
// Packed storage: (((tile*KT + kt)*32 + lane)*16 + e), 32B contiguous per lane.
// ---------------------------------------------------------------------------

// ---- weight packer: fp32 W -> bf16 B-fragment layout (zero-padded) --------
__global__ void pack_w_kernel(const float* __restrict__ W, int wtrans,
                              int K, int N, int KT,
                              __bf16* __restrict__ out, int total)
{
    int i = blockIdx.x*blockDim.x + threadIdx.x;
    if (i >= total) return;
    int e    = i & 15;
    int lane = (i >> 4) & 31;
    int kt   = (i >> 9) % KT;
    int nt   = i / (KT << 9);
    int n    = nt*16 + (lane & 15);
    int half = lane >> 4;
    int j = e >> 1, c = e & 1;
    int k = kt*32 + ((j >> 2) << 4) + (half << 3) + ((j & 3) << 1) + c;
    float v = 0.f;
    if (n < N && k < K) v = wtrans ? W[(size_t)n*K + k] : W[(size_t)k*N + n];
    out[i] = (__bf16)v;
}

// ---- activation packer: fp32 concat(A1,A2) -> bf16 A-fragment layout ------
// Rows padded with zeros up to a multiple of MT*16. Requires K%32==0, K1%32==0.
__global__ void pack_a_kernel(const float* __restrict__ A1, int lda1,
                              const float* __restrict__ A2, int lda2, int K1,
                              int M, int K, int KT,
                              __bf16* __restrict__ out, int total)
{
    int i = blockIdx.x*blockDim.x + threadIdx.x;
    if (i >= total) return;
    int e    = i & 15;
    int lane = (i >> 4) & 31;
    int kt   = (i >> 9) % KT;
    int mt   = i / (KT << 9);
    int m    = mt*16 + (lane & 15);
    int half = lane >> 4;
    int j = e >> 1, c = e & 1;
    int k = kt*32 + ((j >> 2) << 4) + (half << 3) + ((j & 3) << 1) + c;
    float v = 0.f;
    if (m < M && k < K)
        v = (!A2 || k < K1) ? A1[(size_t)m*lda1 + k]
                            : A2[(size_t)m*lda2 + (k - K1)];
    out[i] = (__bf16)v;
}

// ---------------------------------------------------------------------------
// WMMA GEMM on pre-packed bf16 fragments. Per k-step per wave:
//   1x 32B B-fragment load + MT x 32B A-fragment loads + MT x v_wmma.
// ---------------------------------------------------------------------------
__global__ void wmma_gemm_kernel(
    const __bf16* __restrict__ Apk,
    const __bf16* __restrict__ Wpk,
    const float* __restrict__ bias1, const float* __restrict__ bias2,
    const float* __restrict__ res, int ldres,
    float* __restrict__ out, int ldo,
    int M, int N, int K, int act)
{
    const int wid  = threadIdx.x >> 5;
    const int lane = threadIdx.x & 31;
    const int tilesN = (N + 15) >> 4;
    const int MTt    = (M + 15) >> 4;            // 16-row tiles (packed count)
    const int tilesM = (MTt + MT - 1) / MT;
    int tile = blockIdx.x * (blockDim.x >> 5) + wid;
    if (tile >= tilesM * tilesN) return;
    const int tm = tile / tilesN;
    const int tn = tile % tilesN;
    const int half = lane >> 4;
    const int mr   = lane & 15;
    const int KT   = K >> 5;

    v8f acc[MT];
#pragma unroll
    for (int s = 0; s < MT; ++s) acc[s] = (v8f){0.f,0.f,0.f,0.f,0.f,0.f,0.f,0.f};

    const __bf16* wp = Wpk + (((size_t)tn*KT)*32 + lane)*16;
    const __bf16* ap = Apk + ((((size_t)tm*MT)*KT)*32 + lane)*16;

    for (int kt = 0; kt < KT; ++kt){
        v16bf bfrag = *(const v16bf*)wp;
        wp += 512;
#pragma unroll
        for (int s = 0; s < MT; ++s){
            v16bf af = *(const v16bf*)(ap + ((size_t)s*KT + kt)*512);
            acc[s] = __builtin_amdgcn_wmma_f32_16x16x32_bf16(
                         false, af, false, bfrag, (short)0, acc[s], false, false);
        }
    }
#pragma unroll
    for (int s = 0; s < MT; ++s){
#pragma unroll
        for (int i = 0; i < 8; ++i){
            int row = tm*(MT*16) + s*16 + i + (half << 3);   // C layout: M = i + 8*half
            int col = tn*16 + mr;
            if (row < M && col < N){
                float v = acc[s][i];
                if (bias1) v += bias1[col];
                if (bias2) v += bias2[col];
                if (act == 1) v = gelu_t(v);
                if (res) v += res[(size_t)row*ldres + col];
                out[(size_t)row*ldo + col] = v;
            }
        }
    }
}

// ---------------- front-end 7-tap conv + GELU -> g[bc,t,c] -----------------
__global__ void front_conv_kernel(const float* __restrict__ x,
                                  const float* __restrict__ w1,
                                  const float* __restrict__ b1,
                                  float* __restrict__ g, int total)
{
    int i = blockIdx.x*blockDim.x + threadIdx.x;
    if (i >= total) return;
    int c  = i & 127;
    int t  = (i >> 7) % Tv;
    int bc = i / (Tv * Dv);
    const float* xs = x + (size_t)bc * Tv;
    float acc = b1[c];
#pragma unroll
    for (int k = 0; k < 7; ++k){
        int tt = t + k - 3;
        if (tt >= 0 && tt < Tv) acc += xs[tt] * w1[c*7 + k];
    }
    g[i] = gelu_t(acc);
}

// ------- causal depthwise time-conv (DC=4) + SiLU: xz(:, :256) -> xc -------
__global__ void dwconv_silu_kernel(const float* __restrict__ xz,
                                   const float* __restrict__ cw,
                                   const float* __restrict__ cb,
                                   float* __restrict__ xc, int total)
{
    int i = blockIdx.x*blockDim.x + threadIdx.x;
    if (i >= total) return;
    int d  = i & 255;
    int t  = (i >> 8) % Tv;
    int bc = i / (Tv * DIv);
    size_t base = (size_t)bc * Tv;
    float acc = cb[d];
#pragma unroll
    for (int k = 0; k < 4; ++k){
        int tt = t - 3 + k;
        if (tt >= 0) acc += xz[(base + tt)*512 + d] * cw[d*4 + k];
    }
    xc[i] = silu_f(acc);
}

// ---------------- time-reverse copy of (BC,T,128) --------------------------
__global__ void reverse_kernel(const float* __restrict__ src,
                               float* __restrict__ dst, int total)
{
    int i = blockIdx.x*blockDim.x + threadIdx.x;
    if (i >= total) return;
    int d  = i & 127;
    int t  = (i >> 7) % Tv;
    int bc = i / (Tv * Dv);
    dst[(((size_t)bc*Tv + (Tv-1 - t)) << 7) + d] = src[i];
}

// ---------------- selective-scan: dt, SSM recurrence, Dp, SiLU(z) gate -----
__global__ void mamba_scan_kernel(const float* __restrict__ xc,
                                  const float* __restrict__ xdbl,
                                  const float* __restrict__ Wdt,
                                  const float* __restrict__ bdt,
                                  const float* __restrict__ Alog,
                                  const float* __restrict__ Dp,
                                  float* __restrict__ xz)
{
    int bc = blockIdx.x;
    int d  = threadIdx.x;          // 256 threads, one channel each
    __shared__ float sB[16], sC[16], sx[8];
    float Ad[16], hs[16];
#pragma unroll
    for (int n = 0; n < 16; ++n){ Ad[n] = -__expf(Alog[d*16 + n]); hs[n] = 0.f; }
    float wdt[8];
#pragma unroll
    for (int r = 0; r < 8; ++r) wdt[r] = Wdt[r*DIv + d];
    float bdtv = bdt[d], dpv = Dp[d];
    for (int t = 0; t < Tv; ++t){
        size_t row = (size_t)bc*Tv + t;
        __syncthreads();
        if (d < 8)                  sx[d]    = xdbl[row*40 + d];
        else if (d >= 32 && d < 48) sB[d-32] = xdbl[row*40 + 8  + (d-32)];
        else if (d >= 64 && d < 80) sC[d-64] = xdbl[row*40 + 24 + (d-64)];
        __syncthreads();
        float dtp = bdtv;
#pragma unroll
        for (int r = 0; r < 8; ++r) dtp += sx[r]*wdt[r];
        float dt = softplus_f(dtp);
        float xv = xc[row*DIv + d];
        float dx = dt*xv;
        float acc = 0.f;
#pragma unroll
        for (int n = 0; n < 16; ++n){
            hs[n] = hs[n]*__expf(dt*Ad[n]) + dx*sB[n];
            acc  += hs[n]*sC[n];
        }
        float yv = acc + xv*dpv;
        float zv = xz[row*512 + 256 + d];
        yv *= silu_f(zv);
        xz[row*512 + d] = yv;
    }
}

// ---------------- generic LayerNorm: one block per row ---------------------
__global__ void ln_kernel(const float* __restrict__ x,
                          const float* __restrict__ g,
                          const float* __restrict__ b,
                          float* __restrict__ y, int len)
{
    int row = blockIdx.x;
    const float* xr = x + (size_t)row*len;
    float* yr = y + (size_t)row*len;
    __shared__ float s1[256], s2[256];
    float a = 0.f, a2 = 0.f;
    for (int i = threadIdx.x; i < len; i += blockDim.x){
        float v = xr[i]; a += v; a2 += v*v;
    }
    s1[threadIdx.x] = a; s2[threadIdx.x] = a2;
    __syncthreads();
    for (int off = 128; off > 0; off >>= 1){
        if (threadIdx.x < off){
            s1[threadIdx.x] += s1[threadIdx.x + off];
            s2[threadIdx.x] += s2[threadIdx.x + off];
        }
        __syncthreads();
    }
    float m  = s1[0] / (float)len;
    float vv = s2[0] / (float)len - m*m;
    float rs = rsqrtf(vv + 1e-5f);
    for (int i = threadIdx.x; i < len; i += blockDim.x)
        yr[i] = (xr[i] - m)*rs*g[i] + b[i];
}

// ---------------- band gather: feats -> band-major rows --------------------
__global__ void band_gather_kernel(const float* __restrict__ feats,
                                   float* __restrict__ bandf, int total)
{
    int i = blockIdx.x*blockDim.x + threadIdx.x;
    if (i >= total) return;
    int d  = i & 127;
    int j  = (i >> 7) % 9;
    int bc = i / (9*128);
    int w, nk, p, base;
    if (j < 6)      { w = j;   nk = 6; p = 64;  base = 0; }
    else if (j < 8) { w = j-6; nk = 2; p = 192; base = BCv*6; }
    else            { w = 0;   nk = 1; p = 384; base = BCv*8; }
    int pos = (w + 1)*p - 1;
    int outrow = base + bc*nk + w;
    bandf[(size_t)outrow*128 + d] = feats[((size_t)bc*Tv + pos)*128 + d];
}

// ---------------- event-order scatter into (B,C,L=9,D) ---------------------
__global__ void tok_scatter_kernel(const float* __restrict__ bandout,
                                   float* __restrict__ tarr, int total)
{
    int i = blockIdx.x*blockDim.x + threadIdx.x;
    if (i >= total) return;
    int d  = i & 127;
    int m  = (i >> 7) % 9;
    int bc = i / (9*128);
    const int ord[9] = {0,1,2,6,3,4,5,7,8};
    int j = ord[m];
    int w, nk, base;
    if (j < 6)      { w = j;   nk = 6; base = 0; }
    else if (j < 8) { w = j-6; nk = 2; base = BCv*6; }
    else            { w = 0;   nk = 1; base = BCv*8; }
    int src = base + bc*nk + w;
    tarr[i] = bandout[(size_t)src*128 + d];
}

// ---------------- 19x7 depthwise 2-D positional conv, residual add ---------
__global__ void pos_conv_kernel(const float* __restrict__ tarr,
                                const float* __restrict__ pw,
                                const float* __restrict__ pb,
                                float* __restrict__ t2, int total)
{
    int i = blockIdx.x*blockDim.x + threadIdx.x;
    if (i >= total) return;
    int d = i & 127;
    int l = (i >> 7) % 9;
    int c = (i / (9*128)) % 32;
    int b = i / (32*9*128);
    float acc = 0.f;
    for (int ii = 0; ii < 19; ++ii){
        int cc = c + ii - 9;
        if (cc < 0 || cc >= 32) continue;
        for (int jj = 0; jj < 7; ++jj){
            int ll = l + jj - 3;
            if (ll < 0 || ll >= 9) continue;
            acc += tarr[(((size_t)(b*32 + cc))*9 + ll)*128 + d] * pw[d*133 + ii*7 + jj];
        }
    }
    t2[i] = tarr[i] + acc + pb[d];
}

// ---------------- hemisphere permutation gather ----------------------------
__global__ void perm_gather_kernel(const float* __restrict__ t2,
                                   const int* __restrict__ perm,
                                   float* __restrict__ tp, int total)
{
    int i = blockIdx.x*blockDim.x + threadIdx.x;
    if (i >= total) return;
    int d = i & 127;
    int l = (i >> 7) % 9;
    int c = (i / 1152) % 32;
    int b = i / (32*1152);
    int pc = perm[b*32 + c];
    tp[i] = t2[(((size_t)(b*32 + pc))*9 + l)*128 + d];
}

// ---------------- attention logits: ga @ a_w2 + a_b2 -----------------------
__global__ void logits_kernel(const float* __restrict__ ga,
                              const float* __restrict__ w2,
                              const float* __restrict__ b2,
                              float* __restrict__ logits)
{
    int row = blockIdx.x, lane = threadIdx.x;
    float s = 0.f;
    for (int i = lane; i < 200; i += 32) s += ga[(size_t)row*200 + i]*w2[i];
    for (int off = 16; off; off >>= 1) s += __shfl_xor(s, off, 32);
    if (lane == 0) logits[row] = s + b2[0];
}

__global__ void softmax_kernel(const float* __restrict__ logits,
                               float* __restrict__ w)
{
    int b = blockIdx.x, lane = threadIdx.x;   // C == 32 == wave
    float v = logits[b*32 + lane];
    float m = v;
    for (int off = 16; off; off >>= 1) m = fmaxf(m, __shfl_xor(m, off, 32));
    float e = __expf(v - m);
    float s = e;
    for (int off = 16; off; off >>= 1) s += __shfl_xor(s, off, 32);
    w[b*32 + lane] = e / s;
}

__global__ void agg_kernel(const float* __restrict__ flatf,
                           const float* __restrict__ w,
                           float* __restrict__ agg, int total)
{
    int i = blockIdx.x*blockDim.x + threadIdx.x;
    if (i >= total) return;
    int f = i % 1152;
    int b = i / 1152;
    float s = 0.f;
    for (int c = 0; c < 32; ++c)
        s += flatf[((size_t)(b*32 + c))*1152 + f]*w[b*32 + c];
    agg[i] = s;
}

// ---------------------------------------------------------------------------
static inline void gemm(hipStream_t s, __bf16* apk,
                        const float* A1, int lda1,
                        const float* A2, int lda2, int K1,
                        const __bf16* Wpk,
                        const float* b1, const float* b2,
                        const float* res, int ldres,
                        float* out, int ldo, int M, int N, int K, int act)
{
    int KT  = K >> 5;
    int MTt = (M + 15)/16;
    int MTp = ((MTt + MT - 1)/MT)*MT;          // pad row-tiles to wave-tile multiple
    int totalA = MTp*KT*512;
    pack_a_kernel<<<(totalA + 255)/256, 256, 0, s>>>(A1, lda1, A2, lda2, K1,
                                                     M, K, KT, apk, totalA);
    int tiles  = (MTp/MT)*((N + 15)/16);
    int blocks = (tiles + 7)/8;
    wmma_gemm_kernel<<<blocks, 256, 0, s>>>(apk, Wpk, b1, b2, res, ldres,
                                            out, ldo, M, N, K, act);
}

#define L1D(kern, total, ...) \
    kern<<<((total) + 255)/256, 256, 0, stream>>>(__VA_ARGS__)

extern "C" void kernel_launch(void* const* d_in, const int* in_sizes, int n_in,
                              void* d_out, int out_size, void* d_ws, size_t ws_size,
                              hipStream_t stream)
{
    (void)in_sizes; (void)n_in; (void)out_size; (void)ws_size;
    const float* x      = (const float*)d_in[0];
    const int*   perm   = (const int*)  d_in[1];
    const float* sp_w1  = (const float*)d_in[2];
    const float* sp_b1  = (const float*)d_in[3];
    const float* sp_w2  = (const float*)d_in[4];
    const float* sp_b2  = (const float*)d_in[5];
    const float* ln_g   = (const float*)d_in[6];
    const float* ln_b   = (const float*)d_in[7];
    const float* W_in   = (const float*)d_in[8];
    const float* conv_w = (const float*)d_in[9];
    const float* conv_b = (const float*)d_in[10];
    const float* W_x    = (const float*)d_in[11];
    const float* W_dt   = (const float*)d_in[12];
    const float* b_dt   = (const float*)d_in[13];
    const float* A_log  = (const float*)d_in[14];
    const float* Dp     = (const float*)d_in[15];
    const float* W_out  = (const float*)d_in[16];
    const float* b_out  = (const float*)d_in[17];
    const float* fuse_w = (const float*)d_in[18];
    const float* fuse_b = (const float*)d_in[19];
    const float* band_emb = (const float*)d_in[20];
    const float* band_pw  = (const float*)d_in[21];
    const float* band_pb  = (const float*)d_in[22];
    const float* pos_w  = (const float*)d_in[23];
    const float* pos_b  = (const float*)d_in[24];
    const float* hemi_w = (const float*)d_in[25];
    const float* hemi_b = (const float*)d_in[26];
    const float* a_ln_g = (const float*)d_in[27];
    const float* a_ln_b = (const float*)d_in[28];
    const float* a_w1   = (const float*)d_in[29];
    const float* a_b1   = (const float*)d_in[30];
    const float* a_w2   = (const float*)d_in[31];
    const float* a_b2   = (const float*)d_in[32];
    const float* m_ln_g = (const float*)d_in[33];
    const float* m_ln_b = (const float*)d_in[34];
    const float* m_w1   = (const float*)d_in[35];
    const float* m_b1   = (const float*)d_in[36];
    const float* m_w2   = (const float*)d_in[37];
    const float* m_b2   = (const float*)d_in[38];
    float* outp = (float*)d_out;

    const size_t nBT = (size_t)BCv * Tv;       // 49152 rows
    const int    M   = (int)nBT;
    float* fws = (float*)d_ws;
    size_t off = 0;
    auto take = [&](size_t n){ float* p = fws + off; off += n; return p; };
    float* SCR = take(nBT * Dv);    // g, then per-block LN scratch
    float* H   = take(nBT * Dv);
    float* P   = take(nBT * Dv);
    float* HB  = take(nBT * Dv);
    float* XZ  = take(nBT * 512);   // xz; y overwrites xp half; feats reuse
    float* XC  = take(nBT * DIv);   // xc; small buffers reuse post-mamba
    float* XD  = take(nBT * 40);
    float* FE    = XZ;
    float* BANDF = XC;
    float* BANDO = XC + 1*147456;
    float* TARR  = XC + 2*147456;
    float* T2    = XC + 3*147456;
    float* TP    = XC + 4*147456;
    float* FU    = XC + 5*147456;
    float* LNF   = XC + 6*147456;
    float* GA    = XC + 7*147456;
    float* LOG   = GA + 25600;
    float* WSM   = LOG + 128;
    float* AGG   = WSM + 128;
    float* LNA   = AGG + 4608;
    float* MG    = LNA + 4608;

    // ---- bf16 packed-A scratch (max: M=49152, K=256 -> 12.6M bf16) ----
    __bf16* APK = (__bf16*)take(6400000);      // 12.8M bf16 incl. padding

    // ---- bf16 packed-weight arena ----
    __bf16* arena = (__bf16*)(fws + off);
    size_t bfoff = 0;
    auto packW = [&](const float* Wsrc, int wtrans, int K, int N) -> const __bf16* {
        int KT = (K + 31)/32, NT = (N + 15)/16;
        int total = NT*KT*512;
        __bf16* p = arena + bfoff;
        bfoff += (size_t)total;
        pack_w_kernel<<<(total + 255)/256, 256, 0, stream>>>(Wsrc, wtrans, K, N, KT, p, total);
        return p;
    };
    const __bf16* pw_sp2  = packW(sp_w2, 1, Dv, Dv);
    const __bf16* pw_in[4]; const __bf16* pw_x[4]; const __bf16* pw_o[4];
    for (int i = 0; i < 4; ++i){
        pw_in[i] = packW(W_in  + (size_t)i*Dv*512,  0, Dv,  512);
        pw_x[i]  = packW(W_x   + (size_t)i*DIv*40,  0, DIv, 40);
        pw_o[i]  = packW(W_out + (size_t)i*DIv*Dv,  0, DIv, Dv);
    }
    const __bf16* pw_fuse = packW(fuse_w, 0, 2*Dv, Dv);
    const __bf16* pw_band[3];
    for (int k = 0; k < 3; ++k) pw_band[k] = packW(band_pw + (size_t)k*Dv*Dv, 0, Dv, Dv);
    const __bf16* pw_hemi = packW(hemi_w, 0, 2*Dv, Dv);
    const __bf16* pw_a1   = packW(a_w1, 0, 1152, 200);
    const __bf16* pw_m1   = packW(m_w1, 0, 1152, 1024);
    const __bf16* pw_m2   = packW(m_w2, 0, 1024, 768);

    // ---- stage 1: front conv + GELU, pointwise projection ----
    L1D(front_conv_kernel, M*Dv, x, sp_w1, sp_b1, SCR, M*Dv);
    gemm(stream, APK, SCR, Dv, nullptr, 0, Dv, pw_sp2, sp_b2, nullptr,
         nullptr, 0, H, Dv, M, Dv, Dv, 0);
    L1D(reverse_kernel, M*Dv, H, HB, M*Dv);   // hb = h[:, ::-1]

    // ---- stage 2: four Mamba blocks ----
    struct { const float* in; float* out; int i; } plan[4] = {
        { H,  P,  0 }, { P,  H,  1 },   // forward chain  -> result in H
        { HB, P,  2 }, { P,  HB, 3 } }; // backward chain -> result in HB
    for (int s = 0; s < 4; ++s){
        int i = plan[s].i;
        const float* hin = plan[s].in;
        float* hout = plan[s].out;
        ln_kernel<<<M, 256, 0, stream>>>(hin, ln_g + i*Dv, ln_b + i*Dv, SCR, Dv);
        gemm(stream, APK, SCR, Dv, nullptr, 0, Dv, pw_in[i],
             nullptr, nullptr, nullptr, 0, XZ, 512, M, 512, Dv, 0);
        L1D(dwconv_silu_kernel, M*DIv, XZ, conv_w + i*DIv*4, conv_b + i*DIv, XC, M*DIv);
        gemm(stream, APK, XC, DIv, nullptr, 0, DIv, pw_x[i],
             nullptr, nullptr, nullptr, 0, XD, 40, M, 40, DIv, 0);
        mamba_scan_kernel<<<BCv, DIv, 0, stream>>>(XC, XD, W_dt + i*8*DIv,
            b_dt + i*DIv, A_log + i*DIv*16, Dp + i*DIv, XZ);
        gemm(stream, APK, XZ, 512, nullptr, 0, DIv, pw_o[i],
             b_out + i*Dv, nullptr, hin, Dv, hout, Dv, M, Dv, DIv, 0);
    }
    L1D(reverse_kernel, M*Dv, HB, P, M*Dv);   // re-reverse hb -> P

    // ---- stage 3: fuse, bands, pos-conv, hemisphere ----
    gemm(stream, APK, H, Dv, P, Dv, Dv, pw_fuse, fuse_b, nullptr,
         nullptr, 0, FE, Dv, M, Dv, 2*Dv, 0);
    L1D(band_gather_kernel, BCv*9*Dv, FE, BANDF, BCv*9*Dv);
    gemm(stream, APK, BANDF,            Dv, nullptr, 0, Dv, pw_band[0],
         band_pb,        band_emb,        nullptr, 0, BANDO,            Dv, BCv*6, Dv, Dv, 0);
    gemm(stream, APK, BANDF + 768*128,  Dv, nullptr, 0, Dv, pw_band[1],
         band_pb + Dv,   band_emb + Dv,   nullptr, 0, BANDO + 768*128,  Dv, BCv*2, Dv, Dv, 0);
    gemm(stream, APK, BANDF + 1024*128, Dv, nullptr, 0, Dv, pw_band[2],
         band_pb + 2*Dv, band_emb + 2*Dv, nullptr, 0, BANDO + 1024*128, Dv, BCv*1, Dv, Dv, 0);
    L1D(tok_scatter_kernel, BCv*9*Dv, BANDO, TARR, BCv*9*Dv);
    L1D(pos_conv_kernel, BCv*9*Dv, TARR, pos_w, pos_b, T2, BCv*9*Dv);
    L1D(perm_gather_kernel, BCv*9*Dv, T2, perm, TP, BCv*9*Dv);
    gemm(stream, APK, T2, Dv, TP, Dv, Dv, pw_hemi, hemi_b, nullptr,
         nullptr, 0, FU, Dv, BCv*9, Dv, 2*Dv, 0);

    // ---- stage 4: attention pooling + final MLP ----
    ln_kernel<<<BCv, 256, 0, stream>>>(FU, a_ln_g, a_ln_b, LNF, 1152);
    gemm(stream, APK, LNF, 1152, nullptr, 0, 1152, pw_a1, a_b1, nullptr,
         nullptr, 0, GA, 200, BCv, 200, 1152, 1);
    logits_kernel<<<BCv, 32, 0, stream>>>(GA, a_w2, a_b2, LOG);
    softmax_kernel<<<4, 32, 0, stream>>>(LOG, WSM);
    L1D(agg_kernel, 4*1152, FU, WSM, AGG, 4*1152);
    ln_kernel<<<4, 256, 0, stream>>>(AGG, m_ln_g, m_ln_b, LNA, 1152);
    gemm(stream, APK, LNA, 1152, nullptr, 0, 1152, pw_m1, m_b1, nullptr,
         nullptr, 0, MG, 1024, 4, 1024, 1152, 1);
    gemm(stream, APK, MG, 1024, nullptr, 0, 1024, pw_m2, m_b2, nullptr,
         nullptr, 0, outp, 768, 4, 768, 1024, 0);
}
